// Proposal_74406013436498
// MI455X (gfx1250) — matrix operand order repaired and checked by reading, not compiled, not run on hardware
//
#include <hip/hip_runtime.h>
#include <hip/hip_bf16.h>
#include <math.h>

// ---------------- model constants (from reference setup) ----------------
constexpr int  FDIM   = 128;
constexpr int  HID    = 32;
constexpr int  CDIM   = 256;          // 8*HID
constexpr int  DEPTH  = 5;
constexpr int  KHOP   = 3;
constexpr int  HEADS  = 8;
constexpr int  JKH    = 896;          // LSTM hidden
constexpr int  ZW     = 4 * JKH;      // 3584 gate width
constexpr int  FLEX   = 512;
constexpr int  NCLS   = 10;
constexpr int  GATEH  = 672;          // 21*HID
constexpr int  BSZ    = 512;
constexpr int  NPG    = 23;
constexpr long NN     = (long)BSZ * NPG;  // 11776 nodes

typedef __attribute__((ext_vector_type(16))) _Float16 v16h;
typedef __attribute__((ext_vector_type(8)))  _Float16 vh8;
typedef __attribute__((ext_vector_type(8)))  float    v8f;
typedef __attribute__((ext_vector_type(4)))  float    vf4;

// ---------------- workspace arena (float offsets) ----------------
constexpr long OFF_JK    = 0;                               // [5, NN, CDIM]
constexpr long OFF_H     = OFF_JK   + 5 * NN * CDIM;        // [NN, JKH]
constexpr long OFF_CELL  = OFF_H    + NN * JKH;             // [NN, JKH]
constexpr long OFF_ATTF  = OFF_CELL + NN * JKH;             // [5, NN]
constexpr long OFF_ATTB  = OFF_ATTF + 5 * NN;               // [5, NN]
constexpr long OFF_BSUM  = OFF_ATTB + 5 * NN;               // [ZW]
constexpr long OFF_STATS = OFF_BSUM + ZW;                   // [512] col sums/sumsq
constexpr long OFF_DINV  = OFF_STATS + 512;                 // [NN]
constexpr long OFF_GATE  = OFF_DINV + NN;                   // [NN]
constexpr long OFF_POOL  = OFF_GATE + NN;                   // [BSZ, CDIM]
constexpr long OFF_YM    = OFF_POOL + (long)BSZ * CDIM;     // [BSZ, CDIM]
constexpr long OFF_S     = OFF_YM   + (long)BSZ * CDIM;     // overlaid scratch
// phase 1 (GNN) view of scratch
constexpr long S_PRE   = OFF_S;                 // [NN, 128]
constexpr long S_H32   = S_PRE   + NN * FDIM;   // [NN, 32]
constexpr long S_FEATS = S_H32   + NN * HID;    // [NN, 128]
constexpr long S_TAG   = S_FEATS + NN * FDIM;   // [NN, 32]
constexpr long S_G     = S_TAG   + NN * HID;    // [NN, 256]
constexpr long S_AS    = S_G     + NN * CDIM;   // [NN, 8]
constexpr long S_AD    = S_AS    + NN * HEADS;  // [NN, 8]
// phase 2 (LSTM) view: z = [NN, ZW] at OFF_S (169 MB), overlays phase 1
// phase 3 view of scratch (overlays z)
constexpr long S_XG  = OFF_S;                   // [NN, 256]
constexpr long S_T1  = S_XG  + NN * CDIM;       // [NN, 512]
constexpr long S_FF  = S_T1  + NN * FLEX;       // [NN, 256]
constexpr long S_XG2 = S_FF  + NN * CDIM;       // [NN, 256]
constexpr long S_GH  = S_XG2 + NN * CDIM;       // [NN, 672]

// =====================================================================
// WMMA GEMM: C[M,N] = A[M,K] @ B(^T) (+bias) (+ELU) (+=C)
//   BT=true : B is [N,K] row-major (torch weight, x @ W^T)
//   BT=false: B is [K,N] row-major (x @ W)
// Preconditions used by all call sites: K % 32 == 0; all row bases
// 16B-aligned (K, N multiples of 4; arena offsets multiples of 4).
// 256 threads = 8 wave32 waves in 2x4 grid; block tile 64x128;
// wave tile 32x32 = 4 x v_wmma_f32_16x16x32_f16 per K-step.
// =====================================================================
template <bool BT, int MODE, bool ACC, bool BIAS>
__global__ __launch_bounds__(256) void gemm_wmma(
    const float* __restrict__ A, const float* __restrict__ B,
    const float* __restrict__ bias, float* __restrict__ Cm,
    int M, int N, int K) {
  __shared__ _Float16 As[64][32];     // [m][k]
  __shared__ _Float16 Bs[128][32];    // [n][k]
  const int tid  = threadIdx.x;
  const int wave = tid >> 5;
  const int lane = tid & 31;
  const int wr   = wave >> 2;         // 0..1  (row wave)
  const int wc   = wave & 3;          // 0..3  (col wave)
  const int m0   = blockIdx.y * 64;
  const int n0   = blockIdx.x * 128;

  v8f accs[4] = {};                   // [rt*2+ct]

  // staging coordinates (loop-invariant)
  const int ar  = tid >> 2;           // A row 0..63
  const int akk = (tid & 3) * 8;      // A k-chunk
  const int agm = m0 + ar;
  const float* aptr = A + (long)agm * K + akk;

  for (int k0 = 0; k0 < K; k0 += 32) {
    // ---- stage A tile (f32 -> f16), 8 contiguous k per thread ----
    {
      vf4 a0 = {}, a1 = {};
      if (agm < M) {
        const vf4* p = reinterpret_cast<const vf4*>(aptr + k0);
        a0 = p[0];
        a1 = p[1];
        if (k0 + 32 < K) __builtin_prefetch(aptr + k0 + 32, 0, 1);
      }
      vh8 hv;
#pragma unroll
      for (int u = 0; u < 4; ++u) { hv[u] = (_Float16)a0[u];
                                    hv[4 + u] = (_Float16)a1[u]; }
      *reinterpret_cast<vh8*>(&As[ar][akk]) = hv;
    }
    // ---- stage B tile as [n][k], 16 elements per thread ----
    if (BT) {
      const int nn = tid >> 1;              // 0..127
      const int kk = (tid & 1) * 16;        // 0 or 16
      const int gn = n0 + nn;
      vf4 b0 = {}, b1 = {}, b2 = {}, b3 = {};
      if (gn < N) {
        const vf4* p = reinterpret_cast<const vf4*>(B + (long)gn * K + k0 + kk);
        b0 = p[0]; b1 = p[1]; b2 = p[2]; b3 = p[3];
        if (k0 + 32 < K) __builtin_prefetch(B + (long)gn * K + k0 + 32 + kk, 0, 1);
      }
      vh8 h0, h1;
#pragma unroll
      for (int u = 0; u < 4; ++u) { h0[u] = (_Float16)b0[u];
                                    h0[4 + u] = (_Float16)b1[u];
                                    h1[u] = (_Float16)b2[u];
                                    h1[4 + u] = (_Float16)b3[u]; }
      *reinterpret_cast<vh8*>(&Bs[nn][kk])     = h0;
      *reinterpret_cast<vh8*>(&Bs[nn][kk + 8]) = h1;
    } else {
      const int kk  = tid >> 3;             // 0..31
      const int nnb = (tid & 7) * 16;       // 0..112
      const int gk  = k0 + kk;              // always < K (K % 32 == 0)
      const float* src = B + (long)gk * N + n0 + nnb;
      if (n0 + nnb + 16 <= N) {
        const vf4* p = reinterpret_cast<const vf4*>(src);
        vf4 b0 = p[0], b1 = p[1], b2 = p[2], b3 = p[3];
#pragma unroll
        for (int u = 0; u < 4; ++u) {
          Bs[nnb + u][kk]      = (_Float16)b0[u];
          Bs[nnb + 4 + u][kk]  = (_Float16)b1[u];
          Bs[nnb + 8 + u][kk]  = (_Float16)b2[u];
          Bs[nnb + 12 + u][kk] = (_Float16)b3[u];
        }
      } else {                              // N boundary (e.g. N=672, 32)
#pragma unroll
        for (int u = 0; u < 16; ++u) {
          const int gn = n0 + nnb + u;
          Bs[nnb + u][kk] = (_Float16)((gn < N) ? src[u] : 0.0f);
        }
      }
    }
    __syncthreads();

    // ---- fragments ----
    v16h af[2], bf[2];
#pragma unroll
    for (int rt = 0; rt < 2; ++rt) {        // A: lane = row, split-K halves
      const int r  = wr * 32 + rt * 16 + (lane & 15);
      const int kb = (lane >> 4) * 8;
      const vh8 lo = *reinterpret_cast<const vh8*>(&As[r][kb]);
      const vh8 hi = *reinterpret_cast<const vh8*>(&As[r][kb + 16]);
#pragma unroll
      for (int i = 0; i < 8; ++i) { af[rt][i] = lo[i]; af[rt][8 + i] = hi[i]; }
    }
#pragma unroll
    for (int ct = 0; ct < 2; ++ct) {        // B: lane = col, split-K halves
      const int cidx = wc * 32 + ct * 16 + (lane & 15);
      const int kb   = (lane >> 4) * 8;
      const vh8 lo = *reinterpret_cast<const vh8*>(&Bs[cidx][kb]);
      const vh8 hi = *reinterpret_cast<const vh8*>(&Bs[cidx][kb + 16]);
#pragma unroll
      for (int i = 0; i < 8; ++i) { bf[ct][i] = lo[i]; bf[ct][8 + i] = hi[i]; }
    }
#pragma unroll
    for (int rt = 0; rt < 2; ++rt)
#pragma unroll
      for (int ct = 0; ct < 2; ++ct)
        accs[rt * 2 + ct] = __builtin_amdgcn_wmma_f32_16x16x32_f16(
            false, af[rt], false, bf[ct], (short)0, accs[rt * 2 + ct],
            false, false);
    __syncthreads();
  }

  // ---- epilogue (C layout: VGPR r -> row r + 8*(lane/16), col = lane&15) ----
#pragma unroll
  for (int rt = 0; rt < 2; ++rt) {
    const int rbase = m0 + wr * 32 + rt * 16 + (lane >> 4) * 8;
#pragma unroll
    for (int ct = 0; ct < 2; ++ct) {
      const int col = n0 + wc * 32 + ct * 16 + (lane & 15);
      if (col >= N) continue;
      const float bv = BIAS ? bias[col] : 0.0f;
      const v8f a = accs[rt * 2 + ct];
#pragma unroll
      for (int r = 0; r < 8; ++r) {
        const int row = rbase + r;
        if (row >= M) continue;
        float v = a[r] + bv;
        if (MODE == 1) v = v > 0.0f ? v : (expf(v) - 1.0f);   // ELU
        const long idx = (long)row * N + col;
        if (ACC) v += Cm[idx];
        Cm[idx] = v;
      }
    }
  }
}

// =====================================================================
// Elementwise / reduction helper kernels
// =====================================================================
__global__ void k_fill(float* p, float v, long n) {
  long i = (long)blockIdx.x * blockDim.x + threadIdx.x;
  if (i < n) p[i] = v;
}

__global__ void k_addvec(const float* a, const float* b, float* o, int n) {
  int i = blockIdx.x * blockDim.x + threadIdx.x;
  if (i < n) o[i] = a[i] + b[i];
}

// PE + Conv1d(23->23, k=3, pad=1) over feature axis
__global__ void k_pre_conv(const float* __restrict__ x,
                           const float* __restrict__ w,
                           const float* __restrict__ cb,
                           float* __restrict__ out) {
  long idx = (long)blockIdx.x * blockDim.x + threadIdx.x;
  if (idx >= NN * FDIM) return;
  const int p = idx & (FDIM - 1);
  const int o = (int)((idx >> 7) % NPG);
  const int b = (int)(idx / ((long)FDIM * NPG));
  float acc = cb[o];
  const float nl = -9.210340371976184f / (float)NPG;  // -ln(10000)/23
  for (int i = 0; i < NPG; ++i) {
    const float dv = __expf((float)(2 * (i >> 1)) * nl);
#pragma unroll
    for (int k = 0; k < 3; ++k) {
      const int pp = p + k - 1;
      if (pp < 0 || pp >= FDIM) continue;
      const float pe = (i & 1) ? __cosf((float)pp * dv) : __sinf((float)pp * dv);
      acc += w[(o * NPG + i) * 3 + k] * (x[((long)b * NPG + i) * FDIM + pp] + pe);
    }
  }
  out[idx] = acc;
}

// per-column sum / sum-of-squares (one block per column)
__global__ void k_colstats(const float* __restrict__ a, const float* __restrict__ r,
                           int n, int f, float* __restrict__ stats) {
  const int col = blockIdx.x;
  float s = 0.f, s2 = 0.f;
  for (int row = threadIdx.x; row < n; row += blockDim.x) {
    float v = a[(long)row * f + col];
    if (r) v += r[(long)row * f + col];
    s += v; s2 += v * v;
  }
  __shared__ float sh[256], sh2[256];
  sh[threadIdx.x] = s; sh2[threadIdx.x] = s2;
  __syncthreads();
  for (int st = 128; st > 0; st >>= 1) {
    if (threadIdx.x < st) { sh[threadIdx.x] += sh[threadIdx.x + st];
                            sh2[threadIdx.x] += sh2[threadIdx.x + st]; }
    __syncthreads();
  }
  if (threadIdx.x == 0) { stats[col] = sh[0]; stats[f + col] = sh2[0]; }
}

__global__ void k_bn_apply(const float* __restrict__ a, const float* __restrict__ r,
                           float* __restrict__ out, const float* __restrict__ stats,
                           const float* __restrict__ g, const float* __restrict__ b,
                           int n, int f) {
  long idx = (long)blockIdx.x * blockDim.x + threadIdx.x;
  if (idx >= (long)n * f) return;
  const int col = (int)(idx % f);
  const float inv_n = 1.0f / (float)n;
  const float mu  = stats[col] * inv_n;
  const float var = stats[f + col] * inv_n - mu * mu;
  float v = a[idx] + (r ? r[idx] : 0.f);
  v = (v - mu) * rsqrtf(var + 1e-5f);
  if (g) v = v * g[col] + b[col];
  out[idx] = v;
}

__global__ void k_deg(const int* __restrict__ dst, int e, float* __restrict__ deg) {
  int i = blockIdx.x * blockDim.x + threadIdx.x;
  if (i < e) atomicAdd(&deg[dst[i]], 1.0f);
}
__global__ void k_dinv(float* d, long n) {
  long i = (long)blockIdx.x * blockDim.x + threadIdx.x;
  if (i < n) { float v = d[i]; d[i] = v > 0.f ? rsqrtf(v) : 0.f; }
}

__global__ void k_copy_col(const float* __restrict__ src, int ss,
                           float* __restrict__ dst, int ds, long n, int f) {
  long idx = (long)blockIdx.x * blockDim.x + threadIdx.x;
  if (idx >= n * f) return;
  const long r = idx / f; const int cc = (int)(idx % f);
  dst[r * ds + cc] = src[r * ss + cc];
}

// TAG propagation exploiting complete digraph within each 23-node graph:
// out[d] = dinv[d] * (sum_s dinv[s]*in[s] - dinv[d]*in[d])
__global__ __launch_bounds__(256) void k_tag_prop(
    const float* __restrict__ in, float* __restrict__ out,
    const float* __restrict__ dinv, int stride) {
  const int b = blockIdx.x;
  __shared__ float S[HID];
  __shared__ float din[NPG];
  const int tid = threadIdx.x;
  if (tid < NPG) din[tid] = dinv[b * NPG + tid];
  __syncthreads();
  if (tid < HID) {
    float s = 0.f;
    for (int i = 0; i < NPG; ++i)
      s += din[i] * in[((long)b * NPG + i) * stride + tid];
    S[tid] = s;
  }
  __syncthreads();
  for (int o = tid; o < NPG * HID; o += 256) {
    const int d = o / HID, f = o % HID;
    const long ni = (long)b * NPG + d;
    out[ni * stride + f] = din[d] * (S[f] - din[d] * in[ni * stride + f]);
  }
}

// per-(node, head) attention logits for GAT
__global__ void k_gat_alpha(const float* __restrict__ g,
                            const float* __restrict__ wsrc,
                            const float* __restrict__ wdst,
                            float* __restrict__ as, float* __restrict__ ad) {
  long idx = (long)blockIdx.x * blockDim.x + threadIdx.x;
  if (idx >= NN * HEADS) return;
  const long n = idx / HEADS; const int hd = (int)(idx % HEADS);
  float s = 0.f, d = 0.f;
#pragma unroll
  for (int f = 0; f < HID; ++f) {
    const float gv = g[n * CDIM + hd * HID + f];
    s += gv * wsrc[hd * HID + f];
    d += gv * wdst[hd * HID + f];
  }
  as[idx] = s; ad[idx] = d;
}

// dense per-graph GAT attention (complete graph + self loops = all 23 sources)
__global__ __launch_bounds__(256) void k_gat_attn(
    const float* __restrict__ g, const float* __restrict__ as,
    const float* __restrict__ ad, const float* __restrict__ gbias,
    float* __restrict__ out) {
  const int b = blockIdx.x;
  __shared__ float shA[NPG][HEADS];
  __shared__ float shD[NPG][HEADS];
  __shared__ float alpha[NPG][NPG][HEADS];   // [dst][src][head]
  __shared__ float shg[NPG][CDIM];
  const int tid = threadIdx.x;
  for (int o = tid; o < NPG * HEADS; o += 256) {
    shA[o / HEADS][o % HEADS] = as[((long)b * NPG) * HEADS + o];
    shD[o / HEADS][o % HEADS] = ad[((long)b * NPG) * HEADS + o];
  }
  for (int o = tid; o < NPG * CDIM; o += 256)
    shg[o / CDIM][o % CDIM] = g[((long)b * NPG) * CDIM + o];
  __syncthreads();
  for (int o = tid; o < NPG * HEADS; o += 256) {
    const int d = o / HEADS, hd = o % HEADS;
    float av[NPG]; float m = -1e30f;
#pragma unroll
    for (int s = 0; s < NPG; ++s) {
      float a = shA[s][hd] + shD[d][hd];
      a = a > 0.f ? a : 0.2f * a;               // leaky_relu(0.2)
      av[s] = a; m = fmaxf(m, a);
    }
    float sum = 0.f;
#pragma unroll
    for (int s = 0; s < NPG; ++s) { const float e = __expf(av[s] - m);
                                    alpha[d][s][hd] = e; sum += e; }
    const float inv = 1.0f / sum;
#pragma unroll
    for (int s = 0; s < NPG; ++s) alpha[d][s][hd] *= inv;
  }
  __syncthreads();
  for (int o = tid; o < NPG * CDIM; o += 256) {
    const int d = o / CDIM, cc = o % CDIM;
    const int hd = cc / HID;
    float acc = 0.f;
#pragma unroll
    for (int s = 0; s < NPG; ++s) acc += alpha[d][s][hd] * shg[s][cc];
    out[((long)b * NPG + d) * CDIM + cc] = acc + gbias[cc];
  }
}

// LSTM gates from fully-formed z = x@Wih^T + h@Whh^T + (bih+bhh)
__global__ void k_lstm_gate(const float* __restrict__ z, float* __restrict__ c,
                            float* __restrict__ h) {
  long idx = (long)blockIdx.x * blockDim.x + threadIdx.x;
  if (idx >= NN * JKH) return;
  const long row = idx / JKH; const int j = (int)(idx % JKH);
  const float* zr = z + row * ZW;
  const float ig = 1.f / (1.f + __expf(-zr[j]));
  const float fg = 1.f / (1.f + __expf(-zr[JKH + j]));
  const float gg = tanhf(zr[2 * JKH + j]);
  const float og = 1.f / (1.f + __expf(-zr[3 * JKH + j]));
  const float cc = fg * c[idx] + ig * gg;
  c[idx] = cc;
  h[idx] = og * tanhf(cc);
}

// per-row dot product: out[n] = dot(a[n,:len], w) (+ b[0])
__global__ __launch_bounds__(128) void k_rowdot(const float* __restrict__ a, int len,
                                                const float* __restrict__ w,
                                                const float* __restrict__ b,
                                                float* __restrict__ out) {
  const long n = blockIdx.x;
  float s = 0.f;
  for (int j = threadIdx.x; j < len; j += 128) s += a[n * len + j] * w[j];
  __shared__ float sh[128];
  sh[threadIdx.x] = s; __syncthreads();
  for (int st = 64; st > 0; st >>= 1) {
    if (threadIdx.x < st) sh[threadIdx.x] += sh[threadIdx.x + st];
    __syncthreads();
  }
  if (threadIdx.x == 0) out[n] = sh[0] + (b ? b[0] : 0.f);
}

// JK: softmax over layers of (attf+attb+jk_b), weighted sum of jk_seq
__global__ void k_jk_combine(const float* __restrict__ jk,
                             const float* __restrict__ attf,
                             const float* __restrict__ attb,
                             const float* __restrict__ jkb,
                             float* __restrict__ xg) {
  long idx = (long)blockIdx.x * blockDim.x + threadIdx.x;
  if (idx >= NN * CDIM) return;
  const long n = idx / CDIM;
  float a[DEPTH]; float m = -1e30f;
#pragma unroll
  for (int t = 0; t < DEPTH; ++t) {
    a[t] = attf[(long)t * NN + n] + attb[(long)t * NN + n] + jkb[0];
    m = fmaxf(m, a[t]);
  }
  float s = 0.f;
#pragma unroll
  for (int t = 0; t < DEPTH; ++t) { a[t] = __expf(a[t] - m); s += a[t]; }
  const float inv = 1.f / s;
  float acc = 0.f;
#pragma unroll
  for (int t = 0; t < DEPTH; ++t) acc += a[t] * inv * jk[(long)t * NN * CDIM + idx];
  xg[idx] = acc;
}

// GlobalAttention pooling per graph (23 nodes)
__global__ __launch_bounds__(256) void k_pool(const float* __restrict__ xg,
                                              const float* __restrict__ gate,
                                              float* __restrict__ pooled) {
  const int b = blockIdx.x;
  __shared__ float w[NPG];
  __shared__ float inv;
  if (threadIdx.x == 0) {
    float m = -1e30f;
    for (int i = 0; i < NPG; ++i) m = fmaxf(m, gate[b * NPG + i]);
    float s = 0.f;
    for (int i = 0; i < NPG; ++i) { const float e = __expf(gate[b * NPG + i] - m);
                                    w[i] = e; s += e; }
    inv = 1.f / s;
  }
  __syncthreads();
  for (int cc = threadIdx.x; cc < CDIM; cc += 256) {
    float acc = 0.f;
#pragma unroll
    for (int i = 0; i < NPG; ++i) acc += w[i] * xg[((long)b * NPG + i) * CDIM + cc];
    pooled[(long)b * CDIM + cc] = acc * inv;
  }
}

// Conv1d(1->10, k=9, pad 4) over channel axis, mean over out-channels
__global__ __launch_bounds__(256) void k_head_conv(const float* __restrict__ pooled,
                                                   const float* __restrict__ w,
                                                   const float* __restrict__ cb,
                                                   float* __restrict__ ym) {
  const int b = blockIdx.x; const int p = threadIdx.x;  // CDIM == 256 threads
  __shared__ float wm[9]; __shared__ float bm;
  if (threadIdx.x < 9) {
    float s = 0.f;
    for (int o = 0; o < NCLS; ++o) s += w[o * 9 + threadIdx.x];
    wm[threadIdx.x] = s * 0.1f;
  }
  if (threadIdx.x == 9) {
    float s = 0.f; for (int o = 0; o < NCLS; ++o) s += cb[o];
    bm = s * 0.1f;
  }
  __syncthreads();
  float acc = bm;
#pragma unroll
  for (int k = 0; k < 9; ++k) {
    const int pp = p + k - 4;
    if (pp >= 0 && pp < CDIM) acc += wm[k] * pooled[(long)b * CDIM + pp];
  }
  ym[(long)b * CDIM + p] = acc;
}

__global__ __launch_bounds__(32) void k_classify(const float* __restrict__ ym,
                                                 const float* __restrict__ fcw,
                                                 const float* __restrict__ fcb,
                                                 float* __restrict__ out) {
  const int b = blockIdx.x;
  __shared__ float lg[NCLS];
  __shared__ float lse;
  const int t = threadIdx.x;
  if (t < NCLS) {
    float s = fcb[t];
    for (int cc = 0; cc < CDIM; ++cc) s += ym[(long)b * CDIM + cc] * fcw[cc * NCLS + t];
    lg[t] = s;
  }
  __syncthreads();
  if (t == 0) {
    float m = -1e30f;
    for (int j = 0; j < NCLS; ++j) m = fmaxf(m, lg[j]);
    float s = 0.f;
    for (int j = 0; j < NCLS; ++j) s += __expf(lg[j] - m);
    lse = m + __logf(s);
  }
  __syncthreads();
  if (t < NCLS) out[b * NCLS + t] = lg[t] - lse;
}

// =====================================================================
// host-side GEMM dispatch
// =====================================================================
static void gemm(hipStream_t s, const float* A, const float* B, const float* bias,
                 float* Cm, int M, int N, int K, bool bt, bool acc, bool elu) {
  dim3 g((N + 127) / 128, (M + 63) / 64), blk(256);
  if (bt) {
    if (acc) gemm_wmma<true, 0, true, false><<<g, blk, 0, s>>>(A, B, nullptr, Cm, M, N, K);
    else     gemm_wmma<true, 0, false, true><<<g, blk, 0, s>>>(A, B, bias, Cm, M, N, K);
  } else {
    if (elu)        gemm_wmma<false, 1, false, true ><<<g, blk, 0, s>>>(A, B, bias, Cm, M, N, K);
    else if (bias)  gemm_wmma<false, 0, false, true ><<<g, blk, 0, s>>>(A, B, bias, Cm, M, N, K);
    else            gemm_wmma<false, 0, false, false><<<g, blk, 0, s>>>(A, B, nullptr, Cm, M, N, K);
  }
}

static inline int cdiv(long a, int b) { return (int)((a + b - 1) / b); }

extern "C" void kernel_launch(void* const* d_in, const int* in_sizes, int n_in,
                              void* d_out, int out_size, void* d_ws, size_t ws_size,
                              hipStream_t stream) {
  const float* x        = (const float*)d_in[0];
  const int*   edge     = (const int*)d_in[1];
  const float* conv1_w  = (const float*)d_in[4];
  const float* conv1_b  = (const float*)d_in[5];
  const float* bn1_g    = (const float*)d_in[6];
  const float* bn1_b    = (const float*)d_in[7];
  const float* ll1_w    = (const float*)d_in[8];
  const float* ll1_b    = (const float*)d_in[9];
  const float* tag_w    = (const float*)d_in[10];
  const float* tag_b    = (const float*)d_in[11];
  const float* gat_w    = (const float*)d_in[12];
  const float* gat_asrc = (const float*)d_in[13];
  const float* gat_adst = (const float*)d_in[14];
  const float* gat_b    = (const float*)d_in[15];
  const float* lstm_wih[2] = {(const float*)d_in[16], (const float*)d_in[20]};
  const float* lstm_whh[2] = {(const float*)d_in[17], (const float*)d_in[21]};
  const float* lstm_bih[2] = {(const float*)d_in[18], (const float*)d_in[22]};
  const float* lstm_bhh[2] = {(const float*)d_in[19], (const float*)d_in[23]};
  const float* jk_w     = (const float*)d_in[24];
  const float* jk_b     = (const float*)d_in[25];
  const float* fff_w1   = (const float*)d_in[26];
  const float* fff_b1   = (const float*)d_in[27];
  const float* fff_w2   = (const float*)d_in[28];
  const float* fff_b2   = (const float*)d_in[29];
  const float* gate_w1  = (const float*)d_in[30];
  const float* gate_b1  = (const float*)d_in[31];
  const float* gate_w2  = (const float*)d_in[32];
  const float* gate_b2  = (const float*)d_in[33];
  const float* conv2_w  = (const float*)d_in[34];
  const float* conv2_b  = (const float*)d_in[35];
  const float* bn2_g    = (const float*)d_in[36];
  const float* bn2_b    = (const float*)d_in[37];
  const float* fc_w     = (const float*)d_in[38];
  const float* fc_b     = (const float*)d_in[39];

  float* ws = (float*)d_ws;
  float* stats = ws + OFF_STATS;
  const int E = in_sizes[1] / 2;

  // ---------- phase 1: preprocessing ----------
  k_pre_conv<<<cdiv(NN * FDIM, 256), 256, 0, stream>>>(x, conv1_w, conv1_b, ws + S_PRE);
  k_colstats<<<FDIM, 256, 0, stream>>>(ws + S_PRE, nullptr, (int)NN, FDIM, stats);
  k_bn_apply<<<cdiv(NN * FDIM, 256), 256, 0, stream>>>(ws + S_PRE, nullptr, ws + S_PRE,
                                                       stats, bn1_g, bn1_b, (int)NN, FDIM);
  gemm(stream, ws + S_PRE, ll1_w, ll1_b, ws + S_H32, (int)NN, HID, FDIM, false, false, false);

  // gcn_norm degrees -> dinv
  k_fill<<<cdiv(NN, 256), 256, 0, stream>>>(ws + OFF_DINV, 0.f, NN);
  k_deg<<<cdiv(E, 256), 256, 0, stream>>>(edge + E, E, ws + OFF_DINV);
  k_dinv<<<cdiv(NN, 256), 256, 0, stream>>>(ws + OFF_DINV, NN);

  // ---------- phase 1b: DEPTH x (TAGConv + BN + GAT) ----------
  for (int i = 0; i < DEPTH; ++i) {
    k_copy_col<<<cdiv(NN * HID, 256), 256, 0, stream>>>(ws + S_H32, HID,
                                                        ws + S_FEATS, FDIM, NN, HID);
    for (int k = 1; k <= KHOP; ++k)
      k_tag_prop<<<BSZ, 256, 0, stream>>>(ws + S_FEATS + (long)(k - 1) * HID,
                                          ws + S_FEATS + (long)k * HID,
                                          ws + OFF_DINV, FDIM);
    gemm(stream, ws + S_FEATS, tag_w + (long)i * FDIM * HID, tag_b + i * HID,
         ws + S_TAG, (int)NN, HID, FDIM, false, false, false);
    k_colstats<<<HID, 256, 0, stream>>>(ws + S_H32, ws + S_TAG, (int)NN, HID, stats);
    k_bn_apply<<<cdiv(NN * HID, 256), 256, 0, stream>>>(ws + S_H32, ws + S_TAG, ws + S_H32,
                                                        stats, nullptr, nullptr, (int)NN, HID);
    gemm(stream, ws + S_H32, gat_w + (long)i * HID * CDIM, nullptr,
         ws + S_G, (int)NN, CDIM, HID, false, false, false);
    k_gat_alpha<<<cdiv(NN * HEADS, 256), 256, 0, stream>>>(
        ws + S_G, gat_asrc + (long)i * HEADS * HID, gat_adst + (long)i * HEADS * HID,
        ws + S_AS, ws + S_AD);
    k_gat_attn<<<BSZ, 256, 0, stream>>>(ws + S_G, ws + S_AS, ws + S_AD,
                                        gat_b + (long)i * CDIM,
                                        ws + OFF_JK + (long)i * NN * CDIM);
  }

  // ---------- phase 2: JK bi-LSTM (z buffer overlays phase-1 scratch) ----------
  float* z = ws + OFF_S;
  for (int dir = 0; dir < 2; ++dir) {
    k_addvec<<<cdiv(ZW, 256), 256, 0, stream>>>(lstm_bih[dir], lstm_bhh[dir],
                                                ws + OFF_BSUM, ZW);
    k_fill<<<cdiv(NN * JKH, 256), 256, 0, stream>>>(ws + OFF_H, 0.f, NN * JKH);
    k_fill<<<cdiv(NN * JKH, 256), 256, 0, stream>>>(ws + OFF_CELL, 0.f, NN * JKH);
    float* att = ws + (dir ? OFF_ATTB : OFF_ATTF);
    const float* jkw_dir = jk_w + dir * JKH;
    for (int t = 0; t < DEPTH; ++t) {
      const int ts = dir ? (DEPTH - 1 - t) : t;   // backward walks seq reversed
      // z = seq[ts] @ Wih^T + (bih + bhh)
      gemm(stream, ws + OFF_JK + (long)ts * NN * CDIM, lstm_wih[dir], ws + OFF_BSUM,
           z, (int)NN, ZW, CDIM, true, false, false);
      // z += h @ Whh^T   (h==0 at t==0)
      if (t > 0)
        gemm(stream, ws + OFF_H, lstm_whh[dir], nullptr, z, (int)NN, ZW, JKH,
             true, true, false);
      k_lstm_gate<<<cdiv(NN * JKH, 256), 256, 0, stream>>>(z, ws + OFF_CELL, ws + OFF_H);
      // att[dir][ts][n] = dot(h[n], jk_w slice)  -> hidden states never stored
      k_rowdot<<<(int)NN, 128, 0, stream>>>(ws + OFF_H, JKH, jkw_dir, nullptr,
                                            att + (long)ts * NN);
    }
  }

  // ---------- phase 3: JK combine, FFF, pooling, head ----------
  float* xg  = ws + S_XG;  float* t1  = ws + S_T1;  float* ffo = ws + S_FF;
  float* xg2 = ws + S_XG2; float* gh  = ws + S_GH;
  k_jk_combine<<<cdiv(NN * CDIM, 256), 256, 0, stream>>>(ws + OFF_JK, ws + OFF_ATTF,
                                                         ws + OFF_ATTB, jk_b, xg);
  gemm(stream, xg, fff_w1, fff_b1, t1, (int)NN, FLEX, CDIM, false, false, true);   // ELU
  gemm(stream, t1, fff_w2, fff_b2, ffo, (int)NN, CDIM, FLEX, false, false, false);
  k_colstats<<<CDIM, 256, 0, stream>>>(xg, ffo, (int)NN, CDIM, stats);
  k_bn_apply<<<cdiv(NN * CDIM, 256), 256, 0, stream>>>(xg, ffo, xg2, stats,
                                                       nullptr, nullptr, (int)NN, CDIM);
  gemm(stream, xg2, gate_w1, gate_b1, gh, (int)NN, GATEH, CDIM, false, false, false);
  k_rowdot<<<(int)NN, 128, 0, stream>>>(gh, GATEH, gate_w2, gate_b2, ws + OFF_GATE);
  k_pool<<<BSZ, 256, 0, stream>>>(xg2, ws + OFF_GATE, ws + OFF_POOL);
  k_head_conv<<<BSZ, 256, 0, stream>>>(ws + OFF_POOL, conv2_w, conv2_b, ws + OFF_YM);
  k_colstats<<<CDIM, 256, 0, stream>>>(ws + OFF_YM, nullptr, BSZ, CDIM, stats);
  k_bn_apply<<<cdiv((long)BSZ * CDIM, 256), 256, 0, stream>>>(ws + OFF_YM, nullptr,
                                                              ws + OFF_YM, stats,
                                                              bn2_g, bn2_b, BSZ, CDIM);
  k_classify<<<BSZ, 32, 0, stream>>>(ws + OFF_YM, fc_w, fc_b, (float*)d_out);
}
